// GSAttention_7885559955564
// MI455X (gfx1250) — matrix operand order repaired and checked by reading, not compiled, not run on hardware
//
#include <hip/hip_runtime.h>
#include <hip/hip_bf16.h>

// ---------------------------------------------------------------------------
// GSAttention (PVT-style SR attention) for MI455X / gfx1250, wave32 + WMMA.
// All GEMMs via v_wmma_f32_16x16x32_f16; softmax/LN/RoPE in fp32 VALU.
// GEMM: 4 waves/block, each wave owns a 32(M) x 64(N) tile -> 8 WMMA per
// 32-k chunk against 12 b128 fragment loads (B fragments reused by both
// A row-groups).
// ---------------------------------------------------------------------------

typedef _Float16 h8  __attribute__((ext_vector_type(8)));
typedef _Float16 h16 __attribute__((ext_vector_type(16)));
typedef float    f8  __attribute__((ext_vector_type(8)));

#define BATCH   32
#define NSEQ    1024
#define CDIM    768
#define HEADS   12
#define HDIM    64
#define NRED    256          // Nr = 1024 / (SR*SR)
#define KCONV   3072         // 768 * 2 * 2
#define BHTOT   (BATCH*HEADS)
#define SCALE_ATT 0.125f     // 64^-0.5

// ---- WMMA helpers ---------------------------------------------------------

__device__ __forceinline__ f8 wmma_f16(h16 a, h16 b, f8 c) {
  // 8 args: (neg_a, A, neg_b, B, c_mod, C, reuse_a, reuse_b)
  return __builtin_amdgcn_wmma_f32_16x16x32_f16(false, a, false, b,
                                                (short)0, c, false, false);
}

// A-fragment (16x32 f16): caller passes p already offset by lane's k-offset
// (0 for lanes 0-15, +16 for lanes 16-31). Elements 0..15 = k..k+15
// (contiguous, matching the B-fragment).
__device__ __forceinline__ h16 load_a16(const _Float16* p) {
  h8 lo = *(const h8*)(p);
  h8 hi = *(const h8*)(p + 8);
  h16 r;
#pragma unroll
  for (int i = 0; i < 8; ++i) { r[i] = lo[i]; r[i + 8] = hi[i]; }
  return r;
}

// B-fragment (32x16 f16): 16 contiguous k-values for this lane's column.
__device__ __forceinline__ h16 load_b16(const _Float16* p) {
  h8 lo = *(const h8*)(p);
  h8 hi = *(const h8*)(p + 8);
  h16 r;
#pragma unroll
  for (int i = 0; i < 8; ++i) { r[i] = lo[i]; r[i + 8] = hi[i]; }
  return r;
}

// ---- elementwise casts ----------------------------------------------------

__global__ void cast_f32_to_f16(const float* __restrict__ src,
                                _Float16* __restrict__ dst, size_t n) {
  size_t i = (size_t)blockIdx.x * blockDim.x + threadIdx.x;
  if (i < n) dst[i] = (_Float16)src[i];
}

// sr_w (O, I, kh, kw) -> f16 (O, kh*2+kw, I): makes im2col k contiguous.
__global__ void permute_srw(const float* __restrict__ src,
                            _Float16* __restrict__ dst) {
  size_t i = (size_t)blockIdx.x * blockDim.x + threadIdx.x;
  if (i >= (size_t)CDIM * KCONV) return;
  int o = (int)(i / KCONV);
  int rem = (int)(i % KCONV);
  int q = rem / CDIM;      // kh*2+kw
  int c = rem % CDIM;      // input channel
  dst[i] = (_Float16)src[(size_t)o * KCONV + (size_t)c * 4 + q];
}

// ---- generic WMMA GEMM:  C[M x Nc] = A[M x K] * W[Nc x K]^T + bias --------
// Block = 4 waves; wave tile = 32(M) x 64(N); block tile = 128 x 64.
// MODE 0: q-proj  -> f16 out0 laid out (b, h, n, d)
// MODE 1: SR conv -> f32 out0 (M x Nc), A gathered from x16 as 2x2 patches
// MODE 2: kv-proj -> K f16 (b,h,n,d) in out0 ; V f16 transposed (b,h,d,n) out2
// MODE 3: out-proj-> f32 out0 (M x Nc) = final result
template <int MODE>
__global__ __launch_bounds__(128) void gemm16(
    const _Float16* __restrict__ A, const _Float16* __restrict__ Wm,
    const float* __restrict__ bias, void* __restrict__ out0,
    _Float16* __restrict__ out2, int M, int Nc, int K) {
  const int lane = threadIdx.x & 31;
  const int wave = threadIdx.x >> 5;
  const int l15 = lane & 15;
  const int ob = blockIdx.x * 64;
  const int mb = blockIdx.y * 128 + wave * 32;
  const int aoff = (lane < 16) ? 0 : 16;
  const int boff = (lane < 16) ? 0 : 16;

  const _Float16* wrow[4];
#pragma unroll
  for (int nt = 0; nt < 4; ++nt)
    wrow[nt] = Wm + (size_t)(ob + nt * 16 + l15) * K + boff;

  // A row this lane feeds for each of the two 16-row groups
  const int mrow0 = mb + l15;
  const int mrow1 = mb + 16 + l15;

  const _Float16* abase0 = nullptr;
  const _Float16* abase1 = nullptr;
  const _Float16* cpos0[4];
  const _Float16* cpos1[4];
  if constexpr (MODE == 1) {
    // conv rows: m = b*256 + h'*16 + w'; 4 source pixels of the 2x2 patch
    int b0 = mrow0 >> 8, np0 = mrow0 & 255;
    int b1 = mrow1 >> 8, np1 = mrow1 & 255;
    int h0 = np0 >> 4, w0 = np0 & 15;
    int h1 = np1 >> 4, w1 = np1 & 15;
#pragma unroll
    for (int q = 0; q < 4; ++q) {
      int p0 = (2 * h0 + (q >> 1)) * 32 + 2 * w0 + (q & 1);
      int p1 = (2 * h1 + (q >> 1)) * 32 + 2 * w1 + (q & 1);
      cpos0[q] = A + ((size_t)b0 * NSEQ + p0) * CDIM;
      cpos1[q] = A + ((size_t)b1 * NSEQ + p1) * CDIM;
    }
  } else {
    abase0 = A + (size_t)mrow0 * K;
    abase1 = A + (size_t)mrow1 * K;
  }

  f8 acc0[4], acc1[4];
#pragma unroll
  for (int nt = 0; nt < 4; ++nt)
#pragma unroll
    for (int r = 0; r < 8; ++r) { acc0[nt][r] = 0.f; acc1[nt][r] = 0.f; }

  const int nk = K >> 5;
  for (int kc = 0; kc < nk; ++kc) {
    const _Float16 *ap0, *ap1;
    if constexpr (MODE == 1) {
      int kk = kc * 32;                 // 768 % 32 == 0: chunk stays in 1 pixel
      int q = kk / CDIM, rem = kk % CDIM;
      ap0 = cpos0[q] + rem;
      ap1 = cpos1[q] + rem;
    } else {
      ap0 = abase0 + kc * 32;
      ap1 = abase1 + kc * 32;
    }
    h16 a0 = load_a16(ap0 + aoff);
    h16 a1 = load_a16(ap1 + aoff);
#pragma unroll
    for (int nt = 0; nt < 4; ++nt) {
      h16 bf = load_b16(wrow[nt] + kc * 32);
      acc0[nt] = wmma_f16(a0, bf, acc0[nt]);
      acc1[nt] = wmma_f16(a1, bf, acc1[nt]);
    }
  }

  const int rowoff = 8 * (lane >> 4);
#pragma unroll
  for (int rg = 0; rg < 2; ++rg) {
#pragma unroll
    for (int nt = 0; nt < 4; ++nt) {
      int o = ob + nt * 16 + l15;
      float bv = bias[o];
#pragma unroll
      for (int r = 0; r < 8; ++r) {
        int m = mb + rg * 16 + r + rowoff;
        float v = (rg == 0 ? acc0[nt][r] : acc1[nt][r]) + bv;
        if constexpr (MODE == 0) {
          int b_ = m >> 10, n = m & 1023, h = o >> 6, d = o & 63;
          ((_Float16*)out0)[(((size_t)(b_ * HEADS + h)) * NSEQ + n) * HDIM + d] =
              (_Float16)v;
        } else if constexpr (MODE == 1 || MODE == 3) {
          ((float*)out0)[(size_t)m * Nc + o] = v;
        } else {  // MODE 2
          if (o < CDIM) {
            int b_ = m >> 8, n = m & 255, h = o >> 6, d = o & 63;
            ((_Float16*)out0)[(((size_t)(b_ * HEADS + h)) * NRED + n) * HDIM +
                              d] = (_Float16)v;
          } else {
            int o2 = o - CDIM;
            int b_ = m >> 8, n = m & 255, h = o2 >> 6, d = o2 & 63;
            out2[(((size_t)(b_ * HEADS + h)) * HDIM + d) * NRED + n] =
                (_Float16)v;
          }
        }
      }
    }
  }
}

// ---- LayerNorm over C=768, rows = B*Nr ------------------------------------

__global__ __launch_bounds__(256) void ln_kernel(
    const float* __restrict__ xin, const float* __restrict__ g,
    const float* __restrict__ be, _Float16* __restrict__ out) {
  __shared__ float ssum[256], ssq[256];
  const int row = blockIdx.x, t = threadIdx.x;
  const float* xr = xin + (size_t)row * CDIM;
  float v[3], s = 0.f, q = 0.f;
#pragma unroll
  for (int j = 0; j < 3; ++j) {
    v[j] = xr[t + 256 * j];
    s += v[j];
    q += v[j] * v[j];
  }
  ssum[t] = s; ssq[t] = q;
  __syncthreads();
  for (int off = 128; off > 0; off >>= 1) {
    if (t < off) { ssum[t] += ssum[t + off]; ssq[t] += ssq[t + off]; }
    __syncthreads();
  }
  float mu = ssum[0] * (1.f / CDIM);
  float var = ssq[0] * (1.f / CDIM) - mu * mu;
  float rstd = rsqrtf(var + 1e-5f);
#pragma unroll
  for (int j = 0; j < 3; ++j) {
    int i = t + 256 * j;
    out[(size_t)row * CDIM + i] = (_Float16)((v[j] - mu) * rstd * g[i] + be[i]);
  }
}

// ---- 2D rotary (in-place, f16, layout (BH, npos, 64)) ---------------------

__global__ void rope_kernel(_Float16* __restrict__ buf, int npos, int Wdim,
                            float step, float bias) {
  size_t idx = (size_t)blockIdx.x * blockDim.x + threadIdx.x;
  size_t total = (size_t)BHTOT * npos * 32;
  if (idx >= total) return;
  int j = (int)(idx & 31);                 // pair index 0..31
  int n = (int)((idx >> 5) % npos);
  size_t bh = idx / ((size_t)npos * 32);
  int i = j >> 1;
  float freq = __powf(10000.f, -(float)i * (1.f / 16.f));
  float pos = (j & 1) ? (bias + (float)(n / Wdim) * step)
                      : (bias + (float)(n % Wdim) * step);
  float s, c;
  __sincosf(pos * freq, &s, &c);
  size_t base = ((size_t)bh * npos + n) * HDIM + 2 * j;
  float re = (float)buf[base], im = (float)buf[base + 1];
  buf[base]     = (_Float16)(re * c - im * s);
  buf[base + 1] = (_Float16)(re * s + im * c);
}

// ---- attention: per wave 16 q-rows x full 256 keys -------------------------
// Q: (BH, 1024, 64)  K: (BH, 256, 64)  Vt: (BH, 64, 256)  O: (B, N, 768) f16

__global__ __launch_bounds__(128) void attn_kernel(
    const _Float16* __restrict__ Qr, const _Float16* __restrict__ Kr,
    const _Float16* __restrict__ Vt, _Float16* __restrict__ O16) {
  __shared__ __align__(16) _Float16 lds_p[4][16 * NRED];  // 32 KB
  const int lane = threadIdx.x & 31;
  const int wave = threadIdx.x >> 5;
  const int l15 = lane & 15;
  const int aoff = (lane < 16) ? 0 : 16;
  const int boff = (lane < 16) ? 0 : 16;
  const int bh = blockIdx.y;
  const int b = bh / HEADS, h = bh % HEADS;
  const int qbase = blockIdx.x * 64 + wave * 16;

  // Q A-fragments for k=0..31 and k=32..63
  const _Float16* qrow =
      Qr + (((size_t)bh * NSEQ) + qbase + l15) * HDIM;
  h16 qa0 = load_a16(qrow + aoff);
  h16 qa1 = load_a16(qrow + 32 + aoff);

  // S = Q K^T over all 16 key tiles
  f8 acc[16];
#pragma unroll
  for (int t = 0; t < 16; ++t)
#pragma unroll
    for (int r = 0; r < 8; ++r) acc[t][r] = 0.f;

  const _Float16* kb = Kr + (size_t)bh * NRED * HDIM;
#pragma unroll
  for (int t = 0; t < 16; ++t) {
    const _Float16* krow = kb + (size_t)(t * 16 + l15) * HDIM;
    h16 k0 = load_b16(krow + boff);
    h16 k1 = load_b16(krow + 32 + boff);
    acc[t] = wmma_f16(qa0, k0, acc[t]);
    acc[t] = wmma_f16(qa1, k1, acc[t]);
  }

  // row softmax (rows striped: VGPR r + lane-half -> row, 16 lanes per row)
  float rsum[8];
#pragma unroll
  for (int r = 0; r < 8; ++r) {
    float m = acc[0][r];
#pragma unroll
    for (int t = 1; t < 16; ++t) m = fmaxf(m, acc[t][r]);
    m = fmaxf(m, __shfl_xor(m, 1));
    m = fmaxf(m, __shfl_xor(m, 2));
    m = fmaxf(m, __shfl_xor(m, 4));
    m = fmaxf(m, __shfl_xor(m, 8));
    float s = 0.f;
#pragma unroll
    for (int t = 0; t < 16; ++t) {
      float p = __expf((acc[t][r] - m) * SCALE_ATT);
      acc[t][r] = p;
      s += p;
    }
    s += __shfl_xor(s, 1);
    s += __shfl_xor(s, 2);
    s += __shfl_xor(s, 4);
    s += __shfl_xor(s, 8);
    rsum[r] = s;
  }

  // stage P in LDS (row-major 16 x 256) to reshape into A-fragments
  _Float16* pp = &lds_p[wave][0];
  const int rowoff = 8 * (lane >> 4);
#pragma unroll
  for (int t = 0; t < 16; ++t)
#pragma unroll
    for (int r = 0; r < 8; ++r)
      pp[(r + rowoff) * NRED + t * 16 + l15] = (_Float16)acc[t][r];
  __syncthreads();

  // O = P * V
  f8 oacc[4];
#pragma unroll
  for (int dt = 0; dt < 4; ++dt)
#pragma unroll
    for (int r = 0; r < 8; ++r) oacc[dt][r] = 0.f;

  const _Float16* vb = Vt + (size_t)bh * HDIM * NRED;
#pragma unroll
  for (int kk = 0; kk < 8; ++kk) {
    h16 pa = load_a16(pp + l15 * NRED + kk * 32 + aoff);
#pragma unroll
    for (int dt = 0; dt < 4; ++dt) {
      h16 vf = load_b16(vb + (size_t)(dt * 16 + l15) * NRED + kk * 32 + boff);
      oacc[dt] = wmma_f16(pa, vf, oacc[dt]);
    }
  }

  // normalize by rowsum, scatter to (B, N, 768) f16
#pragma unroll
  for (int dt = 0; dt < 4; ++dt) {
    int d = dt * 16 + l15;
#pragma unroll
    for (int r = 0; r < 8; ++r) {
      int q = qbase + r + rowoff;
      float v = oacc[dt][r] / rsum[r];
      O16[(((size_t)b * NSEQ) + q) * CDIM + h * HDIM + d] = (_Float16)v;
    }
  }
}

// ---------------------------------------------------------------------------

extern "C" void kernel_launch(void* const* d_in, const int* in_sizes, int n_in,
                              void* d_out, int out_size, void* d_ws,
                              size_t ws_size, hipStream_t stream) {
  (void)in_sizes; (void)n_in; (void)out_size; (void)ws_size;
  const float* x      = (const float*)d_in[0];
  const float* q_w    = (const float*)d_in[1];
  const float* q_b    = (const float*)d_in[2];
  const float* kv_w   = (const float*)d_in[3];
  const float* kv_b   = (const float*)d_in[4];
  const float* sr_w   = (const float*)d_in[5];
  const float* sr_b   = (const float*)d_in[6];
  const float* ln_g   = (const float*)d_in[7];
  const float* ln_b   = (const float*)d_in[8];
  const float* proj_w = (const float*)d_in[9];
  const float* proj_b = (const float*)d_in[10];
  float* out = (float*)d_out;

  // workspace carve-up (256-B aligned)
  char* ws = (char*)d_ws;
  size_t off = 0;
  auto take = [&](size_t bytes) {
    size_t r = off;
    off += (bytes + 255) & ~(size_t)255;
    return r;
  };
  const size_t E_X16 = (size_t)BATCH * NSEQ * CDIM;      // 25.1M
  const size_t E_QW  = (size_t)CDIM * CDIM;
  const size_t E_KVW = (size_t)2 * CDIM * CDIM;
  const size_t E_SRW = (size_t)CDIM * KCONV;
  const size_t E_PW  = (size_t)CDIM * CDIM;
  const size_t E_QR  = (size_t)BHTOT * NSEQ * HDIM;
  const size_t E_KR  = (size_t)BHTOT * NRED * HDIM;
  const size_t E_XR  = (size_t)BATCH * NRED * CDIM;

  _Float16* x16   = (_Float16*)(ws + take(E_X16 * 2));
  _Float16* qw16  = (_Float16*)(ws + take(E_QW * 2));
  _Float16* kvw16 = (_Float16*)(ws + take(E_KVW * 2));
  _Float16* srw16 = (_Float16*)(ws + take(E_SRW * 2));
  _Float16* pw16  = (_Float16*)(ws + take(E_PW * 2));
  _Float16* Qr    = (_Float16*)(ws + take(E_QR * 2));
  _Float16* Kr    = (_Float16*)(ws + take(E_KR * 2));
  _Float16* Vt    = (_Float16*)(ws + take(E_KR * 2));
  float*    xrpre = (float*)   (ws + take(E_XR * 4));
  _Float16* xr16  = (_Float16*)(ws + take(E_XR * 2));
  _Float16* O16   = x16;  // x16 is dead after the conv GEMM; alias it.

  auto gs = [](size_t n) { return (unsigned)((n + 255) / 256); };

  // 1) casts / weight repack
  cast_f32_to_f16<<<gs(E_X16), 256, 0, stream>>>(x, x16, E_X16);
  cast_f32_to_f16<<<gs(E_QW), 256, 0, stream>>>(q_w, qw16, E_QW);
  cast_f32_to_f16<<<gs(E_KVW), 256, 0, stream>>>(kv_w, kvw16, E_KVW);
  cast_f32_to_f16<<<gs(E_PW), 256, 0, stream>>>(proj_w, pw16, E_PW);
  permute_srw<<<gs(E_SRW), 256, 0, stream>>>(sr_w, srw16);

  // 2) q projection: (32768 x 768) = x16 @ q_w^T -> Qr (b,h,n,d)
  gemm16<0><<<dim3(CDIM / 64, (BATCH * NSEQ) / 128), 128, 0, stream>>>(
      x16, qw16, q_b, (void*)Qr, nullptr, BATCH * NSEQ, CDIM, CDIM);

  // 3) SR conv as GEMM: (8192 x 768), K = 3072 -> xrpre (f32)
  gemm16<1><<<dim3(CDIM / 64, (BATCH * NRED) / 128), 128, 0, stream>>>(
      x16, srw16, sr_b, (void*)xrpre, nullptr, BATCH * NRED, CDIM, KCONV);

  // 4) LayerNorm -> xr16
  ln_kernel<<<BATCH * NRED, 256, 0, stream>>>(xrpre, ln_g, ln_b, xr16);

  // 5) kv projection: (8192 x 1536) -> Kr (b,h,n,d) + Vt (b,h,d,n)
  gemm16<2><<<dim3((2 * CDIM) / 64, (BATCH * NRED) / 128), 128, 0, stream>>>(
      xr16, kvw16, kv_b, (void*)Kr, Vt, BATCH * NRED, 2 * CDIM, CDIM);

  // 6) RoPE: q (W=32, step=1, bias=0); k (W=16, step=2, bias=0.5)
  rope_kernel<<<gs((size_t)BHTOT * NSEQ * 32), 256, 0, stream>>>(
      Qr, NSEQ, 32, 1.0f, 0.0f);
  rope_kernel<<<gs((size_t)BHTOT * NRED * 32), 256, 0, stream>>>(
      Kr, NRED, 16, 2.0f, 0.5f);

  // 7) attention -> O16 (B, N, 768) f16
  attn_kernel<<<dim3(NSEQ / 64, BHTOT), 128, 0, stream>>>(Qr, Kr, Vt, O16);

  // 8) output projection: (32768 x 768) -> d_out (f32)
  gemm16<3><<<dim3(CDIM / 64, (BATCH * NSEQ) / 128), 128, 0, stream>>>(
      O16, pw16, proj_b, (void*)out, nullptr, BATCH * NSEQ, CDIM, CDIM);
}
